// SegnetDownsampleUnit_23742579212939
// MI455X (gfx1250) — matrix operand order repaired and checked by reading, not compiled, not run on hardware
//
#include <hip/hip_runtime.h>
#include <stdint.h>

typedef __attribute__((ext_vector_type(16))) __bf16 v16bf;
typedef __attribute__((ext_vector_type(8)))  float  v8f;

#define C_IN   64
#define C_OUT  128
#define HDIM   224
#define WDIM   224
#define NBATCH 16
#define WTILE  32                 // pixels per block along W
#define HTILE  4                  // output rows per block
#define NWBLK  (WDIM / WTILE)     // 7
#define NHBLK  (HDIM / HTILE)     // 56
#define KSTEPS 18                 // K = 9 taps * 64 ci = 576 = 18 * 32
#define NSLOTS 64
#define CNTF   ((float)(NBATCH * HDIM * WDIM))
#define BN_EPS 1e-5f

#define SROWS   (HTILE + 2)       // staged input rows: 6
#define CIPITCH 72                // padded ci stride (144B = 36 banks -> conflict-free b128)

// ---- workspace layout (float / uint32 units) ----
#define WS_PARTIAL 0                         // NSLOTS * 256 floats (sum[128], sumsq[128] per slot)
#define WS_SB      (NSLOTS * 256)            // 256 floats: interleaved scale/bias per channel
#define WS_WPACK   (WS_SB + 256)             // 8*18*32*8 uint32 packed A fragments
#define NFRAGS     (8 * KSTEPS * 32)

__device__ __forceinline__ uint16_t f2bf(float f) {
  union { float f; uint32_t u; } c; c.f = f;
  uint32_t u = c.u;
  u += 0x7FFFu + ((u >> 16) & 1u);           // round-to-nearest-even
  return (uint16_t)(u >> 16);
}

__global__ void zero_stats_kernel(float* __restrict__ partial) {
  int i = blockIdx.x * blockDim.x + threadIdx.x;
  if (i < NSLOTS * 256) partial[i] = 0.f;
}

// Binarize weights and pre-swizzle into WMMA A fragments in this code's contiguous K order (B matches)
// (16-bit A 16x32: lanes 0-15 -> M=lane, dwords 0..7 = K 0..15;
//  lanes 16-31 -> M=lane-16, K 16..31).
__global__ void pack_weights_kernel(const float* __restrict__ w,
                                    uint32_t* __restrict__ wpack) {
  int t = blockIdx.x * blockDim.x + threadIdx.x;   // fragment id: ((cot*18+s)*32+lane)
  if (t >= NFRAGS) return;
  int lane = t & 31;
  int s    = (t >> 5) % KSTEPS;
  int cot  = t / (32 * KSTEPS);
  int m    = cot * 16 + (lane & 15);
  uint32_t frag[8];
#pragma unroll
  for (int d = 0; d < 8; ++d) {
    uint32_t v = 0;
#pragma unroll
    for (int e = 0; e < 2; ++e) {
      int klocal = ((lane >= 16) ? 16 : 0) + d * 2 + e;
      int K   = s * 32 + klocal;               // K = tap*64 + ci
      int tap = K >> 6;
      int ci  = K & 63;
      int dh  = tap / 3, dw = tap - dh * 3;
      float wv = w[((m * C_IN + ci) * 3 + dh) * 3 + dw];
      uint16_t b = (wv > 0.f) ? 0x3F80u : (wv < 0.f ? 0xBF80u : 0u);  // bf16 +-1 (sign)
      v |= ((uint32_t)b) << (16 * e);
    }
    frag[d] = v;
  }
  uint32_t* dst = wpack + (size_t)t * 8;
#pragma unroll
  for (int d = 0; d < 8; ++d) dst[d] = frag[d];
}

template <bool WRITE_OUT>
__global__ __launch_bounds__(256)
void conv_bn_kernel(const float* __restrict__ x,
                    const uint32_t* __restrict__ wpack,
                    float* __restrict__ partial,
                    const float* __restrict__ sb,
                    float* __restrict__ out) {
  // staged patch: [row(6)][wi(34)][ci padded to 72] bf16  (~29.4 KB)
  __shared__ __align__(16) uint16_t xs[SROWS * 34 * CIPITCH];
  __shared__ float red[2 * C_OUT];

  int b  = blockIdx.x;
  int wb = b % NWBLK;
  int hb = (b / NWBLK) % NHBLK;
  int n  = b / (NWBLK * NHBLK);
  int w0 = wb * WTILE;
  int h0 = hb * HTILE;

  int tid  = threadIdx.x;
  int lane = tid & 31;
  int wave = tid >> 5;

  // ---- stage input patch: global (w-fastest, coalesced) -> LDS (ci-fastest) as bf16 ----
  for (int idx = tid; idx < SROWS * 34 * C_IN; idx += 256) {
    int ci  = idx / (SROWS * 34);
    int rem = idx - ci * (SROWS * 34);
    int row = rem / 34;
    int wi  = rem - row * 34;
    int hh  = h0 + row - 1;
    int ww  = w0 + wi - 1;
    float v = 0.f;
    if ((unsigned)hh < HDIM && (unsigned)ww < WDIM)
      v = x[(((size_t)n * C_IN + ci) * HDIM + hh) * WDIM + ww];
    xs[(row * 34 + wi) * CIPITCH + ci] = f2bf(v);
  }
  if (!WRITE_OUT) {
    if (tid < 2 * C_OUT) red[tid] = 0.f;
  }
  __syncthreads();

  v8f acc[2 * HTILE] = {};
  const int nsel = lane & 15;
  const int kk0  = (lane >> 4) << 4;   // 0 or 16: which K half this lane's B column holds

  const uint4* wp = (const uint4*)wpack + ((size_t)(wave * KSTEPS) * 32 + lane) * 2;

  // unroll-2: enough to overlap global A loads / ds B loads with WMMA,
  // while keeping live VGPRs well under 256 (no spills, no vgpr-msb banks).
#pragma unroll 2
  for (int s = 0; s < KSTEPS; ++s) {
    const int tap = s >> 1;
    const int cib = (s & 1) << 5;
    const int dh  = tap / 3;
    const int dw  = tap - dh * 3;

    union { uint4 u[2]; v16bf v; } A;
    A.u[0] = wp[0];
    A.u[1] = wp[1];
    wp += 64;  // next K-step: 32 lanes * 2 uint4

#pragma unroll
    for (int hr = 0; hr < HTILE; ++hr) {
      const int row  = hr + dh;                      // staged row index
      const int boff = (row * 34 + nsel + dw) * CIPITCH + cib + kk0;
      union { uint4 u[2]; v16bf v; } B0, B1;
      {
        const uint4* p0 = (const uint4*)&xs[boff];
        B0.u[0] = p0[0]; B0.u[1] = p0[1];
        const uint4* p1 = (const uint4*)&xs[boff + 16 * CIPITCH];
        B1.u[0] = p1[0]; B1.u[1] = p1[1];
      }
      acc[2 * hr]     = __builtin_amdgcn_wmma_f32_16x16x32_bf16(
          false, A.v, false, B0.v, (short)0, acc[2 * hr], false, false);
      acc[2 * hr + 1] = __builtin_amdgcn_wmma_f32_16x16x32_bf16(
          false, A.v, false, B1.v, (short)0, acc[2 * hr + 1], false, false);
    }
  }

  const int corow = wave * 16 + ((lane >> 4) << 3);  // M = r + 8*(lane>=16)

  if (!WRITE_OUT) {
#pragma unroll
    for (int r = 0; r < 8; ++r) {
      float sum = 0.f, sq = 0.f;
#pragma unroll
      for (int k = 0; k < 2 * HTILE; ++k) {
        float v = acc[k][r];
        sum += v;
        sq  += v * v;
      }
      atomicAdd(&red[corow + r], sum);
      atomicAdd(&red[C_OUT + corow + r], sq);
    }
    __syncthreads();
    if (tid < 2 * C_OUT) {
      float* slot = partial + (size_t)(blockIdx.x & (NSLOTS - 1)) * (2 * C_OUT);
      atomicAdd(&slot[tid], red[tid]);
    }
  } else {
#pragma unroll
    for (int r = 0; r < 8; ++r) {
      int co = corow + r;
      float scale = sb[2 * co];
      float bias  = sb[2 * co + 1];
#pragma unroll
      for (int hr = 0; hr < HTILE; ++hr) {
        size_t base = (((size_t)n * C_OUT + co) * HDIM + (h0 + hr)) * WDIM + w0;
        out[base + nsel]      = fmaxf(0.f, acc[2 * hr][r]     * scale + bias);
        out[base + 16 + nsel] = fmaxf(0.f, acc[2 * hr + 1][r] * scale + bias);
      }
    }
  }
}

__global__ void finalize_bn_kernel(const float* __restrict__ partial,
                                   const float* __restrict__ gamma,
                                   const float* __restrict__ beta,
                                   float* __restrict__ sb) {
  int co = threadIdx.x;
  if (co >= C_OUT) return;
  float s = 0.f, s2 = 0.f;
  for (int k = 0; k < NSLOTS; ++k) {
    s  += partial[k * (2 * C_OUT) + co];
    s2 += partial[k * (2 * C_OUT) + C_OUT + co];
  }
  float mean  = s / CNTF;
  float var   = s2 / CNTF - mean * mean;
  float scale = gamma[co] * rsqrtf(var + BN_EPS);
  sb[2 * co]     = scale;
  sb[2 * co + 1] = beta[co] - mean * scale;
}

extern "C" void kernel_launch(void* const* d_in, const int* in_sizes, int n_in,
                              void* d_out, int out_size, void* d_ws, size_t ws_size,
                              hipStream_t stream) {
  const float* x      = (const float*)d_in[0];
  const float* weight = (const float*)d_in[1];
  const float* gamma  = (const float*)d_in[2];
  const float* beta   = (const float*)d_in[3];
  float* out = (float*)d_out;
  float* ws  = (float*)d_ws;

  float*    partial = ws + WS_PARTIAL;
  float*    sb      = ws + WS_SB;
  uint32_t* wpack   = (uint32_t*)(ws + WS_WPACK);

  zero_stats_kernel<<<(NSLOTS * 256 + 255) / 256, 256, 0, stream>>>(partial);
  pack_weights_kernel<<<(NFRAGS + 255) / 256, 256, 0, stream>>>(weight, wpack);

  int nblocks = NBATCH * NHBLK * NWBLK;  // 6272 (divisible by NSLOTS=64)
  conv_bn_kernel<false><<<nblocks, 256, 0, stream>>>(x, wpack, partial, sb, out);
  finalize_bn_kernel<<<1, 128, 0, stream>>>(partial, gamma, beta, sb);
  conv_bn_kernel<true><<<nblocks, 256, 0, stream>>>(x, wpack, partial, sb, out);
}
